// LayoutSelfAttention_25993142075548
// MI455X (gfx1250) — matrix operand order repaired and checked by reading, not compiled, not run on hardware
//
#include <hip/hip_runtime.h>
#include <hip/hip_bf16.h>

// Problem constants
#define BB 4
#define SQ 2048
#define DD 256
#define HH 4
#define HD 64
#define NKT (SQ / 64)

typedef __attribute__((ext_vector_type(16))) _Float16     v16h;
typedef __attribute__((ext_vector_type(8)))  _Float16     v8h;
typedef __attribute__((ext_vector_type(8)))  float        v8f;
typedef __attribute__((ext_vector_type(4)))  float        v4f;
typedef __attribute__((ext_vector_type(4)))  int          v4i;
typedef __attribute__((ext_vector_type(8)))  int          v8i;
typedef __attribute__((ext_vector_type(4)))  unsigned int v4u;

// ---------- CDNA5 feature detection / wrappers ----------
#if defined(__has_builtin)
#  if __has_builtin(__builtin_amdgcn_tensor_load_to_lds)
#    define HAVE_TDM 1
#  endif
#  if __has_builtin(__builtin_amdgcn_s_wait_asynccnt)
#    define WAIT_ASYNC(n) __builtin_amdgcn_s_wait_asynccnt((short)(n))
#  endif
#endif
#ifndef HAVE_TDM
#  define HAVE_TDM 0
#endif
#ifndef WAIT_ASYNC
#  define WAIT_ASYNC(n) asm volatile("s_wait_asynccnt %0" ::"n"(n) : "memory")
#endif
#if HAVE_TDM
#  define WAIT_TENSOR(n) __builtin_amdgcn_s_wait_tensorcnt((short)(n))
#  define TILE_ASYNC 9     // 4 K + 4 V + 1 mask async per wave per tile (bias via TDM)
#else
#  define WAIT_TENSOR(n)
#  define TILE_ASYNC 17    // + 8 bias async b128 per wave per tile
#endif

// low 32 bits of a generic pointer into LDS == LDS byte offset (aperture rule)
__device__ __forceinline__ unsigned lds_addr32(const void* p) {
  return (unsigned)(size_t)p;
}

// async copies: LDS[lds_off] <= MEM[gbase + voff], ASYNCcnt-tracked
__device__ __forceinline__ void async_b128(unsigned lds_off, unsigned voff, const void* gbase) {
  asm volatile("global_load_async_to_lds_b128 %0, %1, %2"
               ::"v"(lds_off), "v"(voff), "s"(gbase)
               : "memory");
}
__device__ __forceinline__ void async_b64(unsigned lds_off, unsigned voff, const void* gbase) {
  asm volatile("global_load_async_to_lds_b64 %0, %1, %2"
               ::"v"(lds_off), "v"(voff), "s"(gbase)
               : "memory");
}

#if HAVE_TDM
// 2D TDM: load a 64(cols f32) x 16(rows) tile, row stride SQ, into LDS (contiguous).
__device__ __forceinline__ void tdm_load_bias(const float* rowBase, unsigned byteColOff,
                                              unsigned ldsDst) {
  const unsigned long long ga = (unsigned long long)(size_t)rowBase + (unsigned long long)byteColOff;
  v4u g0;
  g0.x = 1u;                                                   // count=1, user mode
  g0.y = ldsDst;                                               // lds_addr
  g0.z = (unsigned)(ga & 0xFFFFFFFFull);                       // global_addr[31:0]
  g0.w = (unsigned)((ga >> 32) & 0x01FFFFFFull) | (2u << 30);  // addr[56:32] | type=2
  v8i g1;
  g1[0] = (int)(2u << 16);       // data_size = 4B
  g1[1] = (int)(2048u << 16);    // tensor_dim0 = 2048
  g1[2] = (int)(16u << 16);      // tensor_dim1 = 16
  g1[3] = (int)(64u << 16);      // tile_dim0 = 64
  g1[4] = 16;                    // tile_dim1 = 16
  g1[5] = 2048;                  // tensor_dim0_stride = 2048
  g1[6] = 0;
  g1[7] = 0;
  v4i z4 = {0, 0, 0, 0};
  v8i z8 = {0, 0, 0, 0, 0, 0, 0, 0};
  __builtin_amdgcn_tensor_load_to_lds(g0, g1, z4, z4, z8, 0);
}
#endif

// ---------- WMMA helpers ----------
__device__ __forceinline__ v8f wmma_f16(v16h a, v16h b, v8f c) {
  return __builtin_amdgcn_wmma_f32_16x16x32_f16(false, a, false, b, (short)0, c, false, false);
}

// A-fragment (16x32 MxK), row-major source, leading dim ld (elements)
__device__ __forceinline__ v16h load_a_frag(const _Float16* base, int ld, int lane, int kbase) {
  const int m  = lane & 15;
  const int kh = (lane >> 4) & 1;
  const _Float16* p = base + (size_t)m * ld + kbase + kh * 8;
  v16h a;
  *((v4i*)&a)     = *(const v4i*)p;
  *((v4i*)&a + 1) = *(const v4i*)(p + 16);
  return a;
}

// B-fragment (32x16 KxN) from an N-major source src[n][k], leading dim ld
__device__ __forceinline__ v16h load_b_frag(const _Float16* base, int ld, int lane, int kbase) {
  const int n  = lane & 15;
  const int kh = (lane >> 4) & 1;
  const _Float16* p = base + (size_t)n * ld + kbase + kh * 8;
  v16h b;
  *((v4i*)&b)     = *(const v4i*)p;
  *((v4i*)&b + 1) = *(const v4i*)(p + 16);
  return b;
}

// ---------------- Stage 1: conversions ----------------
__global__ __launch_bounds__(256) void cvt_f16_kernel(const float* __restrict__ src,
                                                      _Float16* __restrict__ dst, int n) {
  int i = blockIdx.x * blockDim.x + threadIdx.x;
  if (i < n) dst[i] = (_Float16)src[i];
}

__global__ __launch_bounds__(256) void cvt_mask_kernel(const int* __restrict__ src,
                                                       float* __restrict__ dst, int n) {
  int i = blockIdx.x * blockDim.x + threadIdx.x;
  if (i < n) dst[i] = src[i] ? -1e30f : 0.0f;
}

// ---------------- Stage 2: QKV projection GEMM ----------------
__global__ __launch_bounds__(128) void qkv_gemm_kernel(const _Float16* __restrict__ X,
                                                       const _Float16* __restrict__ W,
                                                       const float* __restrict__ bias,
                                                       _Float16* __restrict__ out,
                                                       float scale, int transposed_v) {
  const int lane = threadIdx.x & 31;
  const int w    = threadIdx.x >> 5;
  const int m0   = blockIdx.x * 64 + w * 16;
  const int n0b  = blockIdx.y * 64;
  const int kh   = lane >> 4;
  const int ln   = lane & 15;

  v16h a[8];
  const _Float16* xrow = X + (size_t)m0 * DD;
#pragma unroll
  for (int kb = 0; kb < 8; ++kb) a[kb] = load_a_frag(xrow, DD, lane, kb * 32);

#pragma unroll
  for (int nt = 0; nt < 4; ++nt) {
    const int n0 = n0b + nt * 16;
    v8f c = {};
    const _Float16* wrow = W + (size_t)n0 * DD;
#pragma unroll
    for (int kb = 0; kb < 8; ++kb) {
      v16h bf = load_b_frag(wrow, DD, lane, kb * 32);
      c = wmma_f16(a[kb], bf, c);
    }
    const int n  = n0 + ln;
    const float bv = bias[n];
    const int h_ = n >> 6;
    const int hd = n & 63;
#pragma unroll
    for (int r = 0; r < 8; ++r) {
      const int g  = m0 + r + kh * 8;
      const int b_ = g >> 11;
      const int s_ = g & (SQ - 1);
      float val = (((float*)&c)[r] + bv) * scale;
      if (!transposed_v)
        out[(((size_t)b_ * HH + h_) * SQ + s_) * HD + hd] = (_Float16)val;
      else
        out[(((size_t)b_ * HH + h_) * HD + hd) * SQ + s_] = (_Float16)val;
    }
  }
}

// ---------------- Stage 3: fused flash attention (transposed dataflow) ----------------
// scores^T = K(A) x Q^T(B); ctx^T = V^T(A) x P^T(B).
// C-layout puts a query in a lane column: softmax = 1 shfl; P^T / bias / output contiguous.
// grid: (S/64, H, B), block 128 (4 waves x 16 query rows)
// dynamic LDS: kbuf 2x8K | vbuf 2x8K | bias 2x16K | P 8K | mask 2x256B
__global__ __launch_bounds__(128) void attn_kernel(const _Float16* __restrict__ q16,
                                                   const _Float16* __restrict__ k16,
                                                   const _Float16* __restrict__ vT16,
                                                   const float* __restrict__ rel,
                                                   const float* __restrict__ fmask,
                                                   float* __restrict__ out) {
  extern __shared__ char smem[];
  _Float16* kbuf = (_Float16*)smem;               // [2][64*64] f16
  _Float16* vbuf = (_Float16*)(smem + 16384);     // [2][64*64] f16
  float*    bbuf = (float*)(smem + 32768);        // [2][64*64] f32
  _Float16* Pbuf = (_Float16*)(smem + 65536);     // [4][16*64] f16 (P^T per wave)
  float*    mbuf = (float*)(smem + 73728);        // [2][64]    f32 mask penalty

  const int lane = threadIdx.x & 31;
  const int w    = threadIdx.x >> 5;
  const int h    = blockIdx.y;
  const int b    = blockIdx.z;
  const int bh   = b * HH + h;
  const int qm0  = blockIdx.x * 64 + w * 16;
  const int kh   = lane >> 4;
  const int ln   = lane & 15;

  // uniform global tile bases
  const _Float16* kTileBase = k16 + (size_t)bh * SQ * HD;
  const _Float16* vTileBase = vT16 + (size_t)bh * HD * SQ;
  const float*    relWave   = rel + ((size_t)bh * SQ + qm0) * SQ;  // wave's 16 q rows
  const float*    fmRow     = fmask + b * SQ;

  // copy-lane mapping for 128B rows (8 lanes/row, 4 rows per b128 instr)
  const int      crow = lane >> 3;
  const unsigned ccol = (unsigned)(lane & 7) * 16u;

  auto issue_tile = [&](int kt, int buf) {
    const unsigned k0 = (unsigned)kt * 64u;
    // K quarter: key rows [w*16, w*16+16)
#pragma unroll
    for (int i = 0; i < 4; ++i) {
      const int row = w * 16 + i * 4 + crow;
      async_b128(lds_addr32(kbuf + buf * 4096 + row * 64) + ccol,
                 (k0 + (unsigned)row) * 128u + ccol, kTileBase);
    }
    // V^T quarter: hd rows [w*16, w*16+16), global row stride SQ f16
#pragma unroll
    for (int i = 0; i < 4; ++i) {
      const int row = w * 16 + i * 4 + crow;
      async_b128(lds_addr32(vbuf + buf * 4096 + row * 64) + ccol,
                 (unsigned)row * (SQ * 2u) + k0 * 2u + ccol, vTileBase);
    }
    // mask penalty tile: 64 f32 = 256B (8B per lane)
    async_b64(lds_addr32(mbuf + buf * 64) + (unsigned)lane * 8u,
              k0 * 4u + (unsigned)lane * 8u, fmRow);
#if HAVE_TDM
    // bias quarter via TDM: 64 cols x 16 rows f32, row stride SQ
    tdm_load_bias(relWave, k0 * 4u, lds_addr32(bbuf + buf * 4096 + w * 16 * 64));
#else
    const int      brow = lane >> 4;
    const unsigned bcol = (unsigned)(lane & 15) * 16u;
#pragma unroll
    for (int i = 0; i < 8; ++i) {
      const int row = i * 2 + brow;
      async_b128(lds_addr32(bbuf + buf * 4096 + (w * 16 + row) * 64) + bcol,
                 (unsigned)row * (SQ * 4u) + k0 * 4u + bcol, relWave);
    }
#endif
  };

  // Q as B-fragments (contraction over d: two depth-32 frags); Q pre-scaled by 1/8
  const _Float16* qbase = q16 + ((size_t)bh * SQ + qm0) * HD;
  const v16h bq0 = load_b_frag(qbase, HD, lane, 0);
  const v16h bq1 = load_b_frag(qbase, HD, lane, 32);

  v8f   acc[4];                       // ctx^T: rows hd (r), col q (lane)
#pragma unroll
  for (int ot = 0; ot < 4; ++ot) acc[ot] = (v8f){};
  float m_q = -1e30f, l_q = 0.0f;     // per-lane (per-query) softmax stats

  _Float16* Pq = Pbuf + w * 16 * 64;  // P^T: [16 q][64 key] row-major

  issue_tile(0, 0);   // prologue

  for (int kt = 0; kt < NKT; ++kt) {
    const int cur = kt & 1;
    if (kt + 1 < NKT) {
      issue_tile(kt + 1, cur ^ 1);
      WAIT_ASYNC(TILE_ASYNC);
      WAIT_TENSOR(1);
    } else {
      WAIT_ASYNC(0);
      WAIT_TENSOR(0);
    }
    __syncthreads();

    const _Float16* kb = kbuf + cur * 4096;
    const _Float16* vb = vbuf + cur * 4096;
    const float*    bp = bbuf + cur * 4096 + (w * 16 + ln) * 64 + kh * 8;  // bias row q
    const float*    pp = mbuf + cur * 64 + kh * 8;                         // mask penalty

    // ---- scores^T = K @ Q^T + bias + penalty ----
    // element (key = nt*16 + kh*8 + r, q = ln) per lane
    float sf[4][8];
#pragma unroll
    for (int nt = 0; nt < 4; ++nt) {
      v8f  c  = {};
      v16h ak = load_a_frag(kb + nt * 16 * 64, 64, lane, 0);
      c = wmma_f16(ak, bq0, c);
      ak = load_a_frag(kb + nt * 16 * 64, 64, lane, 32);
      c = wmma_f16(ak, bq1, c);

      const v4f b0 = *(const v4f*)(bp + nt * 16);
      const v4f b1 = *(const v4f*)(bp + nt * 16 + 4);
      const v4f p0 = *(const v4f*)(pp + nt * 16);
      const v4f p1 = *(const v4f*)(pp + nt * 16 + 4);
#pragma unroll
      for (int r = 0; r < 4; ++r) {
        sf[nt][r]     = ((float*)&c)[r]     + b0[r] + p0[r];
        sf[nt][r + 4] = ((float*)&c)[r + 4] + b1[r] + p1[r];
      }
    }

    // ---- online softmax: one cross-lane hop (lane q <-> q+16) ----
    float mx = sf[0][0];
#pragma unroll
    for (int nt = 0; nt < 4; ++nt)
#pragma unroll
      for (int r = 0; r < 8; ++r) mx = fmaxf(mx, sf[nt][r]);
    mx = fmaxf(mx, __shfl_xor(mx, 16, 32));
    const float mnew = fmaxf(m_q, mx);
    const float corr = __expf(m_q - mnew);
    float ss = 0.0f;
#pragma unroll
    for (int nt = 0; nt < 4; ++nt)
#pragma unroll
      for (int r = 0; r < 8; ++r) {
        float e = __expf(sf[nt][r] - mnew);
        sf[nt][r] = e;
        ss += e;
      }
    ss += __shfl_xor(ss, 16, 32);
    l_q = l_q * corr + ss;
    m_q = mnew;
#pragma unroll
    for (int ot = 0; ot < 4; ++ot) acc[ot] = acc[ot] * corr;

    // ---- P^T to LDS: 8 contiguous f16 per (lane, nt) -> one b128 store each ----
#pragma unroll
    for (int nt = 0; nt < 4; ++nt) {
      v8h pk;
#pragma unroll
      for (int r = 0; r < 8; ++r) pk[r] = (_Float16)sf[nt][r];
      *(v8h*)(Pq + ln * 64 + nt * 16 + kh * 8) = pk;
    }

    // P^T as B-fragments (contraction over 64 keys -> two depth-32 frags)
    const v16h bp0 = load_b_frag(Pq, 64, lane, 0);
    const v16h bp1 = load_b_frag(Pq, 64, lane, 32);

    // ---- ctx^T += V^T(A) @ P^T(B) ----
#pragma unroll
    for (int ot = 0; ot < 4; ++ot) {
      v16h av = load_a_frag(vb + ot * 16 * 64, 64, lane, 0);
      acc[ot] = wmma_f16(av, bp0, acc[ot]);
      av = load_a_frag(vb + ot * 16 * 64, 64, lane, 32);
      acc[ot] = wmma_f16(av, bp1, acc[ot]);
    }

    __syncthreads();   // all waves done with buf[cur] before refill
  }

  // ---- epilogue: normalize, contiguous vectorized stores to [B,S,D] f32 ----
  const float invl = 1.0f / l_q;
  float* orow = out + ((size_t)b * SQ + qm0 + ln) * DD + h * HD + kh * 8;
#pragma unroll
  for (int ot = 0; ot < 4; ++ot) {
    v4f o0, o1;
#pragma unroll
    for (int r = 0; r < 4; ++r) {
      o0[r] = ((float*)&acc[ot])[r] * invl;
      o1[r] = ((float*)&acc[ot])[r + 4] * invl;
    }
    *(v4f*)(orow + ot * 16)     = o0;
    *(v4f*)(orow + ot * 16 + 4) = o1;
  }
}

// ---------------- host launcher ----------------
extern "C" void kernel_launch(void* const* d_in, const int* in_sizes, int n_in,
                              void* d_out, int out_size, void* d_ws, size_t ws_size,
                              hipStream_t stream) {
  const float* hidden = (const float*)d_in[0];
  const int*   mask   = (const int*)d_in[1];
  const float* rel    = (const float*)d_in[2];
  const float* Wq     = (const float*)d_in[3];
  const float* bq     = (const float*)d_in[4];
  const float* Wk     = (const float*)d_in[5];
  const float* bk     = (const float*)d_in[6];
  const float* Wv     = (const float*)d_in[7];
  const float* bv     = (const float*)d_in[8];
  float* out = (float*)d_out;

  char* ws = (char*)d_ws;
  _Float16* hf    = (_Float16*)ws;                  // 4 MB hidden f16
  _Float16* wqf   = (_Float16*)(ws + (4u << 20));   // 128 KB each
  _Float16* wkf   = wqf + DD * DD;
  _Float16* wvf   = wkf + DD * DD;
  _Float16* q16   = (_Float16*)(ws + (5u << 20));   // [b,h,s,hd]
  _Float16* k16   = (_Float16*)(ws + (9u << 20));   // [b,h,s,hd]
  _Float16* vT16  = (_Float16*)(ws + (13u << 20));  // [b,h,hd,s]
  float*    fmsk  = (float*)(ws + (17u << 20));     // 32 KB additive mask

  const int nh = BB * SQ * DD;
  const int nw = DD * DD;
  const int nm = BB * SQ;
  cvt_f16_kernel<<<(nh + 255) / 256, 256, 0, stream>>>(hidden, hf, nh);
  cvt_f16_kernel<<<(nw + 255) / 256, 256, 0, stream>>>(Wq, wqf, nw);
  cvt_f16_kernel<<<(nw + 255) / 256, 256, 0, stream>>>(Wk, wkf, nw);
  cvt_f16_kernel<<<(nw + 255) / 256, 256, 0, stream>>>(Wv, wvf, nw);
  cvt_mask_kernel<<<(nm + 255) / 256, 256, 0, stream>>>(mask, fmsk, nm);

  dim3 gg(BB * SQ / 64, DD / 64);
  qkv_gemm_kernel<<<gg, 128, 0, stream>>>(hf, wqf, bq, q16, 0.125f, 0);
  qkv_gemm_kernel<<<gg, 128, 0, stream>>>(hf, wkf, bk, k16, 1.0f, 0);
  qkv_gemm_kernel<<<gg, 128, 0, stream>>>(hf, wvf, bv, vT16, 1.0f, 1);

  dim3 ga(SQ / 64, HH, BB);
  attn_kernel<<<ga, 128, 74240, stream>>>(q16, k16, vT16, rel, fmsk, out);
}
